// TaskLoad_34978213658814
// MI455X (gfx1250) — compile-verified, matrix operand order checked by reading, not run on hardware
//
#include <hip/hip_runtime.h>
#include <hip/hip_bf16.h>
#include <math.h>

// ---------------- problem constants (from reference) ----------------
#define E_EDGES 400000
#define N_USER  50000
#define N_SRV   500
#define N_SRV_P 512
#define HDIM    128
#define ALPHA_L 0.2f
#define CEILDIV(a,b) (((a)+(b)-1)/(b))

typedef __bf16 bf16;
typedef __attribute__((ext_vector_type(16))) __bf16 v16bf;
typedef __attribute__((ext_vector_type(8)))  __bf16 v8bf;
typedef __attribute__((ext_vector_type(8)))  float  v8f;

// ======================================================================
// WMMA GEMM, full-width N=128 strip per wave, weights staged in LDS:
//   Y[M,128] = act( A[M,K](bf16, optionally row-gathered) @ W[128,K]^T + b )
// - One wave computes a 16x128 output strip: A fragment loaded ONCE per
//   K-step (one global b128 pair), reused across 8 N-tiles / 8 WMMAs.
// - W (<=73.7KB bf16) is cooperatively copied into a 64KB LDS tile in
//   K-chunks of 256: every wave in the block shares it via ds_load_b128,
//   keeping the global/VMEM path free for the HBM-bound A stream and
//   cutting L2 weight traffic 8x (once per block instead of per wave).
// Fragment layouts per ISA 7.12.2:
//   A: row = lane&15, K-half = lane>>4 ; B: weight row n, same K pattern;
//   C/D: col = lane&15, rows = (lane>>4)*8 + i.
// ======================================================================
#define KCHUNK 256
template<int ACT, int OBF>
__global__ void k_gemm128(const bf16* __restrict__ A, int lda,
                          const int* __restrict__ rowidx,
                          const bf16* __restrict__ W, int ldw,
                          const float* __restrict__ bias,
                          float* __restrict__ oF, bf16* __restrict__ oH,
                          int ldo, int coloff, int M, int K)
{
    __shared__ bf16 sW[128 * KCHUNK];            // 64 KB weight tile

    const int lane = threadIdx.x & 31;
    const int mtiles = (M + 15) >> 4;
    int mt = blockIdx.x * (blockDim.x >> 5) + (threadIdx.x >> 5);
    const bool active = (mt < mtiles);           // no early return: barriers below
    if (!active) mt = mtiles - 1;                // clamp for safe addresses
    const int r0   = mt << 4;
    const int half = lane >> 4;
    const int sub  = lane & 15;
    int r = r0 + sub; if (r > M - 1) r = M - 1;  // clamp loads, mask stores
    const int ar = rowidx ? rowidx[r] : r;       // fused gather (L2-resident table)
    const bf16* Ap = A + (size_t)ar * lda + (half << 3);

    v8f acc[8];
#pragma unroll
    for (int n = 0; n < 8; ++n) acc[n] = (v8f){0.f,0.f,0.f,0.f,0.f,0.f,0.f,0.f};

    for (int kc = 0; kc < K; kc += KCHUNK) {
        const int chunk = (K - kc < KCHUNK) ? (K - kc) : KCHUNK;   // mult of 32
        // ---- cooperative stage: W[:, kc..kc+chunk) -> sW (row stride KCHUNK)
        {
            const int vchunk = chunk >> 3;                 // v8bf per row
            const int vtot   = 128 * vchunk;
            for (int i = threadIdx.x; i < vtot; i += blockDim.x) {
                int n  = i / vchunk, kv = i % vchunk;
                *(v8bf*)(sW + n * KCHUNK + (kv << 3)) =
                    *(const v8bf*)(W + (size_t)n * ldw + kc + (kv << 3));
            }
        }
        __syncthreads();
        // ---- accumulate over this K-chunk
        for (int k0 = kc; k0 < kc + chunk; k0 += 32) {
            v8bf alo = *(const v8bf*)(Ap + k0);
            v8bf ahi = *(const v8bf*)(Ap + k0 + 16);
            v16bf av;
#pragma unroll
            for (int i = 0; i < 8; ++i) { av[i] = alo[i]; av[i + 8] = ahi[i]; }
            const bf16* Wl = sW + (size_t)sub * KCHUNK + (half << 3) + (k0 - kc);
#pragma unroll
            for (int n = 0; n < 8; ++n) {
                const bf16* Wp = Wl + (size_t)(n << 4) * KCHUNK;
                v8bf blo = *(const v8bf*)(Wp);
                v8bf bhi = *(const v8bf*)(Wp + 16);
                v16bf bv;
#pragma unroll
                for (int i = 0; i < 8; ++i) { bv[i] = blo[i]; bv[i + 8] = bhi[i]; }
                acc[n] = __builtin_amdgcn_wmma_f32_16x16x32_bf16(
                             false, av, false, bv, (short)0, acc[n], false, false);
            }
        }
        __syncthreads();
    }

#pragma unroll
    for (int n = 0; n < 8; ++n) {
        const int col = (n << 4) + sub;
        const float bz = bias ? bias[col] : 0.f;
#pragma unroll
        for (int i = 0; i < 8; ++i) {
            int rr = r0 + (half << 3) + i;
            if (active && rr < M) {
                float v = acc[n][i] + bz;
                if (ACT == 1) v = v > 0.f ? v : 0.f;
                if (ACT == 2) v = v > 0.f ? v : ALPHA_L * v;
                size_t o = (size_t)rr * ldo + coloff + col;
                if (OBF) oH[o] = (bf16)v; else oF[o] = v;
            }
        }
    }
}

// N==1 linear (attention/head final layers): one wave per edge.
// Each lane loads one contiguous v8bf (128-bit) chunk -> the wave reads the
// whole bf16 row as a coalesced 512B burst (K<=256), then shfl-reduces.
template<int ACT>
__global__ void k_dot(const bf16* __restrict__ A, int lda, int K,
                      const float* __restrict__ w, const float* __restrict__ b,
                      float* __restrict__ out, int E)
{
    int lane = threadIdx.x & 31;
    int e = blockIdx.x * (blockDim.x >> 5) + (threadIdx.x >> 5);
    if (e >= E) return;
    const bf16* row = A + (size_t)e * lda;
    float s = 0.f;
    for (int kb = lane << 3; kb < K; kb += 32 * 8) {
        v8bf x = *(const v8bf*)(row + kb);
#pragma unroll
        for (int i = 0; i < 8; ++i) s += (float)x[i] * w[kb + i];
    }
    for (int off = 16; off > 0; off >>= 1) s += __shfl_down(s, off, 32);
    if (lane == 0) {
        float v = s + b[0];
        if (ACT == 1) v = v > 0.f ? v : 0.f;
        if (ACT == 2) v = v > 0.f ? v : ALPHA_L * v;
        out[e] = v;
    }
}

// ---------------- elementwise / data-movement kernels ----------------
__global__ void k_fillf(float* p, float v, int n)
{ int i = blockIdx.x*blockDim.x+threadIdx.x; if (i < n) p[i] = v; }

__global__ void k_fillu(unsigned* p, unsigned v, int n)
{ int i = blockIdx.x*blockDim.x+threadIdx.x; if (i < n) p[i] = v; }

// fp32 [rows,K] -> bf16 [rows,Kp], zero-padded K..Kp (weights + encoder input)
__global__ void k_cast_pad(bf16* dst, const float* src, int rows, int K, int Kp)
{
    int i = blockIdx.x*blockDim.x+threadIdx.x;
    if (i >= rows * Kp) return;
    int r = i / Kp, k = i % Kp;
    dst[i] = (k < K) ? (bf16)src[(size_t)r * K + k] : (bf16)0.f;
}

__global__ void k_build_msgin(bf16* dst, const bf16* xd, const int* idx,
                              const float* ea, int E)
{   // [E,160] = [x_i(128, gathered) | eattr(1) | 0-pad(31)]
    int i = blockIdx.x*blockDim.x+threadIdx.x;
    if (i >= E * 160) return;
    int e = i / 160, c = i % 160;
    bf16 v;
    if (c < 128)       v = xd[(size_t)idx[e] * 128 + c];
    else if (c == 128) v = (bf16)ea[e];
    else               v = (bf16)0.f;
    dst[i] = v;
}

__global__ void k_build_relin(bf16* dst, const bf16* xs, const int* idx,
                              const float* out, const float* ea, int E)
{   // [E,288] = [x_j(128, gathered) | out(128) | eattr(1) | 0-pad(31)]
    int i = blockIdx.x*blockDim.x+threadIdx.x;
    if (i >= E * 288) return;
    int e = i / 288, c = i % 288;
    bf16 v;
    if (c < 128)       v = xs[(size_t)idx[e] * 128 + c];
    else if (c < 256)  v = (bf16)out[(size_t)e * 128 + (c - 128)];
    else if (c == 256) v = (bf16)ea[e];
    else               v = (bf16)0.f;
    dst[i] = v;
}

__global__ void k_build_updin(bf16* dst, const float* aggr, const bf16* xd, int n)
{   // [n,256] = [aggr(128) | x_dst(128)]
    int i = blockIdx.x*blockDim.x+threadIdx.x;
    if (i >= n * 256) return;
    int r = i / 256, c = i % 256;
    dst[i] = (c < 128) ? (bf16)aggr[(size_t)r * 128 + c]
                       : xd[(size_t)r * 128 + (c - 128)];
}

__global__ void k_scale_rows(float* m, const float* a, int E, int H)
{   // out = attention * message, in place
    int i = blockIdx.x*blockDim.x+threadIdx.x;
    if (i >= E * H) return;
    m[i] *= a[i / H];
}

__global__ void k_aggr(float* aggr, const float* out, const int* idx, int E, int H)
{
    int i = blockIdx.x*blockDim.x+threadIdx.x;
    if (i >= E * H) return;
    int e = i / H, h = i % H;
    atomicAdd(&aggr[(size_t)idx[e] * H + h], out[i]);
}

// ---------------- segment softmax (stable, atomic) ----------------
__device__ __forceinline__ unsigned f2ord(float f)
{   unsigned b = __float_as_uint(f);
    return (b & 0x80000000u) ? ~b : (b | 0x80000000u); }
__device__ __forceinline__ float ord2f(unsigned u)
{   unsigned b = (u & 0x80000000u) ? (u ^ 0x80000000u) : ~u;
    return __uint_as_float(b); }

__global__ void k_seg_max(const float* a, const int* idx, unsigned* segm, int E)
{
    int e = blockIdx.x*blockDim.x+threadIdx.x;
    if (e >= E) return;
    atomicMax(&segm[idx[e]], f2ord(a[e]));
}
__global__ void k_seg_expsum(const float* a, const int* idx, const unsigned* segm,
                             float* ev, float* segs, int E)
{
    int e = blockIdx.x*blockDim.x+threadIdx.x;
    if (e >= E) return;
    float m = ord2f(segm[idx[e]]);
    float v = __expf(a[e] - m);
    ev[e] = v;
    atomicAdd(&segs[idx[e]], v);
}
__global__ void k_seg_norm(float* dst, const float* ev, const float* segs,
                           const int* idx, int E)
{
    int e = blockIdx.x*blockDim.x+threadIdx.x;
    if (e >= E) return;
    dst[e] = ev[e] / (segs[idx[e]] + 1e-16f);
}

// ======================================================================
// host orchestration
// ======================================================================
struct CIdx { int wqW,wqB,wrW,wrB,a0W,a0B,a1W,a1B,m0W,m0B,m1W,m1B,
                  r0W,r0B,r1W,r1B,uW,uB; };

extern "C" void kernel_launch(void* const* d_in, const int* in_sizes, int n_in,
                              void* d_out, int out_size, void* d_ws, size_t ws_size,
                              hipStream_t stream)
{
    (void)out_size; (void)ws_size;
    // ---- input mapping: params pytree flattened with sorted dict keys ----
    const float* leaf[84];
    const float *x_user, *x_server, *ea_s2u, *ea_u2s;
    const int *uidx, *sidx;
    if (n_in >= 90 && in_sizes[0] == E_EDGES) {
        // full-alphabetical flatten of the top-level dict
        ea_s2u = (const float*)d_in[0];  ea_u2s = (const float*)d_in[1];
        for (int i = 0; i < 84; ++i) leaf[i] = (const float*)d_in[2 + i];
        sidx = (const int*)d_in[86];     uidx = (const int*)d_in[87];
        x_server = (const float*)d_in[88]; x_user = (const float*)d_in[89];
    } else {
        // insertion order: params..., x_user, x_server, ea_s2u, ea_u2s, uidx, sidx
        for (int i = 0; i < 84; ++i) leaf[i] = (const float*)d_in[i];
        x_user = (const float*)d_in[84]; x_server = (const float*)d_in[85];
        ea_s2u = (const float*)d_in[86]; ea_u2s  = (const float*)d_in[87];
        uidx   = (const int*)d_in[88];   sidx    = (const int*)d_in[89];
    }

    // ---- workspace carving ----
    size_t off = 0;
    auto carve = [&](size_t bytes) -> char* {
        char* p = (char*)d_ws + off;
        off = (off + bytes + 255) & ~(size_t)255;
        return p;
    };
    bf16*  wbuf  = (bf16*)carve((size_t)2*1024*1024*sizeof(bf16)); // padded bf16 weights
    bf16*  xuA   = (bf16*)carve((size_t)N_USER*HDIM*2);
    bf16*  xuB   = (bf16*)carve((size_t)N_USER*HDIM*2);
    bf16*  xsA   = (bf16*)carve((size_t)N_SRV_P*HDIM*2);
    bf16*  xsB   = (bf16*)carve((size_t)N_SRV_P*HDIM*2);
    bf16*  encIn = (bf16*)carve((size_t)N_USER*32*2);
    bf16*  eScr  = (bf16*)carve((size_t)E_EDGES*288*2);   // att-cat(256)/msgin(160)/relin(288)
    bf16*  bufH  = (bf16*)carve((size_t)E_EDGES*HDIM*2);
    bf16*  rS2U  = (bf16*)carve((size_t)E_EDGES*HDIM*2);
    bf16*  rU2S  = (bf16*)carve((size_t)E_EDGES*HDIM*2);
    bf16*  updIn = (bf16*)carve((size_t)N_USER*256*2);
    float* mOut  = (float*)carve((size_t)E_EDGES*HDIM*4);
    float* avec  = (float*)carve((size_t)E_EDGES*4);
    float* evec  = (float*)carve((size_t)E_EDGES*4);
    unsigned* segm = (unsigned*)carve((size_t)N_USER*4);
    float* segs  = (float*)carve((size_t)N_USER*4);
    float* aggr  = (float*)carve((size_t)N_USER*HDIM*4);

    // ---- weight conversion (fp32 -> bf16, K zero-padded to mult-of-32) ----
    size_t wcur = 0;
    auto cvt = [&](const float* w, int N, int K, int Kp) -> bf16* {
        bf16* dst = wbuf + wcur; wcur += (size_t)N * Kp;
        int tot = N * Kp;
        k_cast_pad<<<CEILDIV(tot,256),256,0,stream>>>(dst, w, N, K, Kp);
        return dst;
    };

    // N is always 128; one wave per 16x128 strip, 8 waves/block.
    auto gemm = [&](const bf16* A, int lda, const int* rowidx,
                    const bf16* W, int K, const float* bias,
                    float* oF, bf16* oH, int ldo, int coloff, int M, int act) {
        dim3 blk(256), grd(CEILDIV(CEILDIV(M,16),8));
        if (oH) { if (act) k_gemm128<1,1><<<grd,blk,0,stream>>>(A,lda,rowidx,W,K,bias,oF,oH,ldo,coloff,M,K);
                  else     k_gemm128<0,1><<<grd,blk,0,stream>>>(A,lda,rowidx,W,K,bias,oF,oH,ldo,coloff,M,K); }
        else    { if (act) k_gemm128<1,0><<<grd,blk,0,stream>>>(A,lda,rowidx,W,K,bias,oF,oH,ldo,coloff,M,K);
                  else     k_gemm128<0,0><<<grd,blk,0,stream>>>(A,lda,rowidx,W,K,bias,oF,oH,ldo,coloff,M,K); }
    };

    auto segsm = [&](float* logits, const int* idx, int nseg, float* dst) {
        k_fillu<<<CEILDIV(nseg,256),256,0,stream>>>(segm, 0x007FFFFFu, nseg); // f2ord(-inf)
        k_fillf<<<CEILDIV(nseg,256),256,0,stream>>>(segs, 0.f, nseg);
        k_seg_max   <<<CEILDIV(E_EDGES,256),256,0,stream>>>(logits, idx, segm, E_EDGES);
        k_seg_expsum<<<CEILDIV(E_EDGES,256),256,0,stream>>>(logits, idx, segm, evec, segs, E_EDGES);
        k_seg_norm  <<<CEILDIV(E_EDGES,256),256,0,stream>>>(dst, evec, segs, idx, E_EDGES);
    };

    auto conv = [&](const CIdx& ci, bool s2u,
                    const bf16* xsrc, const bf16* xdstIn, bf16* xdstOut,
                    const int* srcIdx, const int* dstIdx, const float* eattr,
                    int n_src, int n_dst, bf16* rOut) {
        bf16* Wq  = cvt(leaf[ci.wqW], 128, 128, 128);
        bf16* Wr  = cvt(leaf[ci.wrW], 128, 128, 128);
        bf16* Wa0 = s2u ? nullptr : cvt(leaf[ci.a0W], 128, 256, 256);
        bf16* Wm0 = cvt(leaf[ci.m0W], 128, 129, 160);
        bf16* Wm1 = s2u ? cvt(leaf[ci.m1W], 128, 128, 128) : nullptr;
        bf16* Wr0 = cvt(leaf[ci.r0W], 128, 257, 288);
        bf16* Wr1 = cvt(leaf[ci.r1W], 128, 128, 128);
        bf16* Wu  = cvt(leaf[ci.uW],  128, 256, 256);

        // q | r -> eScr[E,256]  (endpoint gather fused into the GEMM A-loads)
        gemm(xdstIn, 128, dstIdx, Wq, 128, leaf[ci.wqB], nullptr, eScr, 256, 0,   E_EDGES, 0);
        gemm(xsrc,   128, srcIdx, Wr, 128, leaf[ci.wrB], nullptr, eScr, 256, 128, E_EDGES, 0);
        // attention logits -> avec
        if (s2u) {
            k_dot<2><<<CEILDIV(E_EDGES,8),256,0,stream>>>(eScr, 256, 256,
                    leaf[ci.a0W], leaf[ci.a0B], avec, E_EDGES);
        } else {
            gemm(eScr, 256, nullptr, Wa0, 256, leaf[ci.a0B], nullptr, bufH, 128, 0, E_EDGES, 1);
            k_dot<2><<<CEILDIV(E_EDGES,8),256,0,stream>>>(bufH, 128, 128,
                    leaf[ci.a1W], leaf[ci.a1B], avec, E_EDGES);
        }
        segsm(avec, srcIdx, n_src, avec);   // softmax over SOURCE segments
        // message MLP
        k_build_msgin<<<CEILDIV(E_EDGES*160,256),256,0,stream>>>(eScr, xdstIn, dstIdx, eattr, E_EDGES);
        if (s2u) {
            gemm(eScr, 160, nullptr, Wm0, 160, leaf[ci.m0B], nullptr, bufH, 128, 0, E_EDGES, 1);
            gemm(bufH, 128, nullptr, Wm1, 128, leaf[ci.m1B], mOut, nullptr, 128, 0, E_EDGES, 1);
        } else {
            gemm(eScr, 160, nullptr, Wm0, 160, leaf[ci.m0B], mOut, nullptr, 128, 0, E_EDGES, 1);
        }
        k_scale_rows<<<CEILDIV(E_EDGES*HDIM,256),256,0,stream>>>(mOut, avec, E_EDGES, HDIM);
        // relation MLP -> rOut (bf16, feeds heads)
        k_build_relin<<<CEILDIV(E_EDGES*288,256),256,0,stream>>>(eScr, xsrc, srcIdx, mOut, eattr, E_EDGES);
        gemm(eScr, 288, nullptr, Wr0, 288, leaf[ci.r0B], nullptr, bufH, 128, 0, E_EDGES, 1);
        gemm(bufH, 128, nullptr, Wr1, 128, leaf[ci.r1B], nullptr, rOut, 128, 0, E_EDGES, 1);
        // aggregate + node update
        k_fillf<<<CEILDIV(n_dst*HDIM,256),256,0,stream>>>(aggr, 0.f, n_dst*HDIM);
        k_aggr<<<CEILDIV(E_EDGES*HDIM,256),256,0,stream>>>(aggr, mOut, dstIdx, E_EDGES, HDIM);
        k_build_updin<<<CEILDIV(n_dst*256,256),256,0,stream>>>(updIn, aggr, xdstIn, n_dst);
        gemm(updIn, 256, nullptr, Wu, 256, leaf[ci.uB], nullptr, xdstOut, 128, 0, n_dst, 1);
    };

    // ---- encoders ----
    k_cast_pad<<<CEILDIV(N_USER*32,256),256,0,stream>>>(encIn, x_user, N_USER, 8, 32);
    { bf16* W0 = cvt(leaf[80], 128, 8, 32);  bf16* W1 = cvt(leaf[82], 128, 128, 128);
      gemm(encIn, 32, nullptr, W0, 32, leaf[81], nullptr, xuB, 128, 0, N_USER, 1);
      gemm(xuB, 128, nullptr, W1, 128, leaf[83], nullptr, xuA, 128, 0, N_USER, 1); }
    k_cast_pad<<<CEILDIV(N_SRV*32,256),256,0,stream>>>(encIn, x_server, N_SRV, 8, 32);
    { bf16* W0 = cvt(leaf[72], 128, 8, 32);  bf16* W1 = cvt(leaf[74], 128, 128, 128);
      gemm(encIn, 32, nullptr, W0, 32, leaf[73], nullptr, xsB, 128, 0, N_SRV, 1);
      gemm(xsB, 128, nullptr, W1, 128, leaf[75], nullptr, xsA, 128, 0, N_SRV, 1); }

    // ---- message-passing layers (double-buffered node features) ----
    auto s2uIdx = [](int l){ int b = 4 + 32*l;
        return CIdx{b,b+1,b+2,b+3,b+4,b+5,-1,-1,b+6,b+7,b+8,b+9,b+10,b+11,b+12,b+13,b+14,b+15}; };
    auto u2sIdx = [](int l){ int b = 20 + 32*l;
        return CIdx{b,b+1,b+2,b+3,b+4,b+5,b+6,b+7,b+8,b+9,-1,-1,b+10,b+11,b+12,b+13,b+14,b+15}; };

    bf16 *xuIn = xuA, *xsIn = xsA, *xuOut = xuB, *xsOut = xsB;
    for (int l = 0; l < 2; ++l) {
        CIdx cs = s2uIdx(l), cu = u2sIdx(l);
        conv(cs, true,  xsIn, xuIn, xuOut, sidx, uidx, ea_s2u, N_SRV,  N_USER, rS2U);
        conv(cu, false, xuIn, xsIn, xsOut, uidx, sidx, ea_u2s, N_USER, N_SRV,  rU2S);
        bf16* t;
        t = xuIn; xuIn = xuOut; xuOut = t;
        t = xsIn; xsIn = xsOut; xsOut = t;
    }

    // ---- heads: seg-softmax(leaky(L1(relu(L0(attr))))) -> d_out ----
    float* dout = (float*)d_out;
    auto head = [&](int w0, int b0, int w1, int b1, bf16* attr,
                    const int* idx, int nseg, float* dst) {
        bf16* W0 = cvt(leaf[w0], 128, 128, 128);
        gemm(attr, 128, nullptr, W0, 128, leaf[b0], nullptr, bufH, 128, 0, E_EDGES, 1);
        k_dot<2><<<CEILDIV(E_EDGES,8),256,0,stream>>>(bufH, 128, 128,
                leaf[w1], leaf[b1], avec, E_EDGES);
        segsm(avec, idx, nseg, dst);
    };
    head(76, 77, 78, 79, rU2S, uidx, N_USER, dout);                // task
    head(68, 69, 70, 71, rU2S, uidx, N_USER, dout + E_EDGES);     // power
    head( 0,  1,  2,  3, rS2U, sidx, N_SRV,  dout + 2*E_EDGES);   // comp
}
